// Attention_523986010726
// MI455X (gfx1250) — matrix operand order repaired and checked by reading, not compiled, not run on hardware
//
#include <hip/hip_runtime.h>
#include <hip/hip_bf16.h>
#include <stdint.h>

// ---------------- problem constants (from reference) ----------------
#define B_    2
#define S_    2048
#define DIM_  4096
#define HQ_   32
#define HKV_  8
#define HD_   128
#define NREP_ 4

// padded LDS row strides (bf16 elems): 16B-aligned and bank-conflict-free
#define PAD32  40    // for 32-wide tiles  (80 B rows: bank = 20*l16 mod 64, distinct)
#define PAD128 136   // for 128-wide tiles (272 B rows: bank = 17*l16 mod 64, distinct)

typedef __attribute__((ext_vector_type(16))) __bf16 v16bf;
typedef __attribute__((ext_vector_type(8)))  float  v8f;

union Frag16 {
    v16bf v;
    unsigned short u[16];
    unsigned int   w[8];
    uint4          q[2];
};

__device__ __forceinline__ unsigned short f32_to_bf16(float f) {
    unsigned int x = __float_as_uint(f);
    x = (x + 0x7FFFu + ((x >> 16) & 1u)) >> 16;   // round-to-nearest-even
    return (unsigned short)x;
}
__device__ __forceinline__ float bf16_to_f32(unsigned short u) {
    return __uint_as_float((unsigned int)u << 16);
}

// stage 16 contiguous elements into LDS as bf16 (overloads: f32 converts, bf16 copies)
__device__ __forceinline__ void stage_row16(unsigned short* dst, const float* src) {
    #pragma unroll
    for (int i = 0; i < 16; i += 4) {
        float4 f = *(const float4*)(src + i);
        uint2 p;
        p.x = (unsigned)f32_to_bf16(f.x) | ((unsigned)f32_to_bf16(f.y) << 16);
        p.y = (unsigned)f32_to_bf16(f.z) | ((unsigned)f32_to_bf16(f.w) << 16);
        *(uint2*)(dst + i) = p;
    }
}
__device__ __forceinline__ void stage_row16(unsigned short* dst, const unsigned short* src) {
    ((uint4*)dst)[0] = ((const uint4*)src)[0];
    ((uint4*)dst)[1] = ((const uint4*)src)[1];
}

__device__ __forceinline__ void store_c(float* C, size_t off, float v)          { C[off] = v; }
__device__ __forceinline__ void store_c(unsigned short* C, size_t off, float v) { C[off] = f32_to_bf16(v); }

// CDNA5 async global->LDS copy (ASYNCcnt-tracked), per ISA 15.18.3 opcode 98
__device__ __forceinline__ void async_ld_b128(unsigned short* lds_dst,
                                              const unsigned short* gsrc) {
    unsigned lds = (unsigned)(uintptr_t)lds_dst;   // low 32 bits of generic addr = LDS addr
    asm volatile("global_load_async_to_lds_b128 %0, %1, off"
                 :: "v"(lds), "v"(gsrc) : "memory");
}
__device__ __forceinline__ void wait_async0() {
    asm volatile("s_wait_asynccnt 0" ::: "memory");
}

// ---------------- 1. WMMA GEMM: C[M,N] = A[M,K] * B[N,K]^T + bias ----------------
// A: f32 or bf16 (converted while staging). B: f32 weights. C: f32 or bf16.
// block tile 128(M) x 256(N); 8 waves in 2(M) x 4(N); wave tile 64x64 = 4x4 WMMA tiles.
template <typename TA, typename TC>
__global__ __launch_bounds__(256)
void gemm_nt_k(const TA* __restrict__ A, const float* __restrict__ Bm,
               const float* __restrict__ bias, TC* __restrict__ C,
               int M, int N, int K) {
    __shared__ unsigned short At[128 * PAD32];
    __shared__ unsigned short Bt[256 * PAD32];

    const int tid  = threadIdx.x;
    const int lane = tid & 31;
    const int wv   = tid >> 5;
    const int hf   = lane >> 4;
    const int l16  = lane & 15;
    const int waveM = wv & 1;      // 0..1
    const int waveN = wv >> 1;     // 0..3
    const int m0 = blockIdx.y * 128;
    const int n0 = blockIdx.x * 256;

    v8f acc[4][4];
    for (int i = 0; i < 4; i++)
        for (int j = 0; j < 4; j++)
            for (int r = 0; r < 8; r++) acc[i][j][r] = 0.f;

    const int ar = tid >> 1;            // A stage: row, 2 threads/row
    const int ac = (tid & 1) * 16;      // A stage: col half

    for (int k0 = 0; k0 < K; k0 += 32) {
        // ---- stage A(128x32) + B(256x32) into LDS (f32 -> bf16 convert in flight) ----
        stage_row16(At + ar * PAD32 + ac, A + (size_t)(m0 + ar) * K + k0 + ac);
        stage_row16(Bt + tid * PAD32,      Bm + (size_t)(n0 + tid) * K + k0);
        stage_row16(Bt + tid * PAD32 + 16, Bm + (size_t)(n0 + tid) * K + k0 + 16);
        __syncthreads();

        // ---- A fragments (16x32 bf16): M=l16, K pairs per VGPR (ISA 7.12.2) ----
        Frag16 af[4];
        #pragma unroll
        for (int i = 0; i < 4; i++) {
            const unsigned short* base = At + (waveM * 64 + i * 16 + l16) * PAD32;
            #pragma unroll
            for (int v = 0; v < 8; v++) {
                int kk = hf * 16 + v * 2;
                af[i].w[v] = *(const unsigned int*)(base + kk);
            }
        }
        // ---- B fragments (32x16 bf16): N=l16, K = hf*16 + i contiguous ----
        Frag16 bfr[4];
        #pragma unroll
        for (int j = 0; j < 4; j++) {
            const uint4* base =
                (const uint4*)(Bt + (waveN * 64 + j * 16 + l16) * PAD32 + hf * 16);
            bfr[j].q[0] = base[0];
            bfr[j].q[1] = base[1];
        }
        #pragma unroll
        for (int i = 0; i < 4; i++)
            #pragma unroll
            for (int j = 0; j < 4; j++)
                acc[i][j] = __builtin_amdgcn_wmma_f32_16x16x32_bf16(
                    false, af[i].v, false, bfr[j].v, (short)0, acc[i][j], false, false);
        __syncthreads();
    }

    // ---- epilogue: D element r -> M = r + hf*8, N = l16 ----
    #pragma unroll
    for (int i = 0; i < 4; i++) {
        #pragma unroll
        for (int j = 0; j < 4; j++) {
            int n = n0 + waveN * 64 + j * 16 + l16;
            float bb = bias ? bias[n] : 0.f;
            #pragma unroll
            for (int r = 0; r < 8; r++) {
                int m = m0 + waveM * 64 + i * 16 + r + hf * 8;
                store_c(C, (size_t)m * N + n, acc[i][j][r] + bb);
            }
        }
    }
}

// ---------------- 2. RoPE + scatter Q,K to (B,H,S,HD) bf16 ----------------
__global__ void rope_scatter_k(const unsigned short* __restrict__ Qp,  // [B*S, HQ*HD] bf16
                               const unsigned short* __restrict__ Kp,  // [B*S, HKV*HD]
                               const float* __restrict__ cosT,         // [S, HD]
                               const float* __restrict__ sinT,
                               unsigned short* __restrict__ Qh,        // [B,HQ,S,HD]
                               unsigned short* __restrict__ Kh) {      // [B,HKV,S,HD]
    int bs = blockIdx.x;
    int b = bs / S_, s = bs % S_;
    const float* crow = cosT + (size_t)s * HD_;
    const float* srow = sinT + (size_t)s * HD_;

    for (int n = threadIdx.x; n < HQ_ * HD_; n += blockDim.x) {
        int h = n >> 7, d = n & 127;
        float x  = bf16_to_f32(Qp[(size_t)bs * (HQ_ * HD_) + n]);
        float xo = bf16_to_f32(Qp[(size_t)bs * (HQ_ * HD_) + (n ^ 64)]);
        float rot = (d < 64) ? -xo : xo;
        Qh[(((size_t)b * HQ_ + h) * S_ + s) * HD_ + d] =
            f32_to_bf16(x * crow[d] + rot * srow[d]);
    }
    for (int n = threadIdx.x; n < HKV_ * HD_; n += blockDim.x) {
        int h = n >> 7, d = n & 127;
        float x  = bf16_to_f32(Kp[(size_t)bs * (HKV_ * HD_) + n]);
        float xo = bf16_to_f32(Kp[(size_t)bs * (HKV_ * HD_) + (n ^ 64)]);
        float rot = (d < 64) ? -xo : xo;
        Kh[(((size_t)b * HKV_ + h) * S_ + s) * HD_ + d] =
            f32_to_bf16(x * crow[d] + rot * srow[d]);
    }
}

// ---------------- 3. V transpose: [B*S, HKV*HD] -> (B,HKV,HD,S) ----------------
// 32x32 tiles through LDS, coalesced on both sides. grid (S/32, HD/32, B*HKV).
__global__ __launch_bounds__(256)
void transpose_v_k(const unsigned short* __restrict__ Vb,
                   unsigned short* __restrict__ VhT) {
    __shared__ unsigned short tile[32][33];
    const int bh = blockIdx.z;
    const int b = bh / HKV_, hk = bh % HKV_;
    const int s0 = blockIdx.x * 32;
    const int d0 = blockIdx.y * 32;
    const int tr = threadIdx.x >> 5;   // 0..7
    const int tc = threadIdx.x & 31;

    #pragma unroll
    for (int i = 0; i < 32; i += 8) {
        int s = s0 + tr + i;
        tile[tr + i][tc] =
            Vb[((size_t)b * S_ + s) * (HKV_ * HD_) + hk * HD_ + d0 + tc];
    }
    __syncthreads();
    #pragma unroll
    for (int i = 0; i < 32; i += 8) {
        int d = d0 + tr + i;
        VhT[(((size_t)b * HKV_ + hk) * HD_ + d) * S_ + s0 + tc] = tile[tc][tr + i];
    }
}

// ---------------- 4. causal flash attention (async double-buffered K/V) ----------------
// grid: (S/128, B*HQ); block 256 = 8 waves; wave w owns 16 query rows.
__global__ __launch_bounds__(256)
void flash_attn_k(const unsigned short* __restrict__ Qh,
                  const unsigned short* __restrict__ Kh,   // (B,HKV,S,HD)
                  const unsigned short* __restrict__ VhT,  // (B,HKV,HD,S)
                  unsigned short* __restrict__ Oh) {       // [B,S,HQ*HD] bf16
    __shared__ unsigned short Kt[2][32 * PAD128];   // [key][d]
    __shared__ unsigned short Vt[2][128 * PAD32];   // [d][key] (transposed tile)
    __shared__ unsigned short Pt[8 * 16 * PAD32];

    const int bh = blockIdx.y;
    const int b = bh / HQ_, h = bh % HQ_;
    const int hk = h / NREP_;
    const int tid = threadIdx.x, lane = tid & 31, wv = tid >> 5;
    const int hf = lane >> 4, l16 = lane & 15;
    const int qblk = blockIdx.x * 128;
    const int q0 = qblk + wv * 16;

    const unsigned short* Qbase = Qh  + ((size_t)b * HQ_  + h)  * S_ * HD_;
    const unsigned short* Kbase = Kh  + ((size_t)b * HKV_ + hk) * S_ * HD_;
    const unsigned short* Vbase = VhT + ((size_t)b * HKV_ + hk) * HD_ * S_;

    // staging coords: K tile 32(key)x128(d), V tile 128(d)x32(key); 16 elems/thread
    const int kr = tid >> 3, kc = (tid & 7) * 16;   // K: row 0..31, col 0..112
    const int vr = tid >> 1, vc = (tid & 1) * 16;   // V: row 0..127, col 0 or 16

    // Q fragments (A operand, 4 K-steps of 32 cover HD=128), kept in VGPRs
    Frag16 qf[4];
    {
        const unsigned short* qrow = Qbase + (size_t)(q0 + l16) * HD_;
        #pragma unroll
        for (int kk = 0; kk < 4; kk++)
            #pragma unroll
            for (int v = 0; v < 8; v++) {
                int kpos = kk * 32 + hf * 16 + v * 2;
                qf[kk].w[v] = *(const unsigned int*)(qrow + kpos);
            }
    }

    v8f Oacc[8];
    for (int dt = 0; dt < 8; dt++)
        for (int r = 0; r < 8; r++) Oacc[dt][r] = 0.f;
    float mrow[8], lrow[8];
    for (int r = 0; r < 8; r++) { mrow[r] = -3.0e38f; lrow[r] = 0.f; }
    const float scale = 0.08838834764831845f;  // 1/sqrt(128)

    const int nIter = (qblk + 128) / 32;  // uniform causal bound for the block

    // prologue: async-fill buffer 0 (keys 0..31)
    {
        const unsigned short* ks = Kbase + (size_t)kr * HD_ + kc;
        const unsigned short* vs = Vbase + (size_t)vr * S_ + vc;
        unsigned short* kd = &Kt[0][kr * PAD128 + kc];
        unsigned short* vd = &Vt[0][vr * PAD32 + vc];
        async_ld_b128(kd, ks);  async_ld_b128(kd + 8, ks + 8);
        async_ld_b128(vd, vs);  async_ld_b128(vd + 8, vs + 8);
    }

    for (int jb = 0; jb < nIter; jb++) {
        const int k0 = jb * 32;
        const int cur = jb & 1;
        wait_async0();          // this wave's fills of buf[cur] done
        __syncthreads();        // all waves' fills done; all done reading buf[1-cur]

        if (jb + 1 < nIter) {   // prefetch next key block into the other buffer
            const int kn = (jb + 1) * 32;
            const unsigned short* ks = Kbase + (size_t)(kn + kr) * HD_ + kc;
            const unsigned short* vs = Vbase + (size_t)vr * S_ + kn + vc;
            unsigned short* kd = &Kt[1 - cur][kr * PAD128 + kc];
            unsigned short* vd = &Vt[1 - cur][vr * PAD32 + vc];
            async_ld_b128(kd, ks);  async_ld_b128(kd + 8, ks + 8);
            async_ld_b128(vd, vs);  async_ld_b128(vd + 8, vs + 8);
        }

        // ---- S = Q K^T : two 16x16 score tiles ----
        v8f sc[2];
        for (int t = 0; t < 2; t++)
            for (int r = 0; r < 8; r++) sc[t][r] = 0.f;
        #pragma unroll
        for (int kk = 0; kk < 4; kk++) {
            #pragma unroll
            for (int t = 0; t < 2; t++) {
                Frag16 kf;
                const uint4* base =
                    (const uint4*)(&Kt[cur][(t * 16 + l16) * PAD128 + kk * 32 + hf * 16]);
                kf.q[0] = base[0];
                kf.q[1] = base[1];
                sc[t] = __builtin_amdgcn_wmma_f32_16x16x32_bf16(
                    false, qf[kk].v, false, kf.v, (short)0, sc[t], false, false);
            }
        }

        // ---- online softmax (row reductions across 16-lane halves, wave32-safe) ----
        float pv0[8], pv1[8], alpha[8];
        #pragma unroll
        for (int r = 0; r < 8; r++) {
            int qp  = q0 + r + hf * 8;
            int kp0 = k0 + l16;
            int kp1 = k0 + 16 + l16;
            float s0 = sc[0][r] * scale + ((kp0 <= qp) ? 0.f : -1.0e9f);
            float s1 = sc[1][r] * scale + ((kp1 <= qp) ? 0.f : -1.0e9f);
            float mx = fmaxf(s0, s1);
            for (int m = 1; m < 16; m <<= 1) mx = fmaxf(mx, __shfl_xor(mx, m, 32));
            float mnew = fmaxf(mrow[r], mx);
            float a  = __expf(mrow[r] - mnew);
            float p0 = __expf(s0 - mnew);
            float p1 = __expf(s1 - mnew);
            float ps = p0 + p1;
            for (int m = 1; m < 16; m <<= 1) ps += __shfl_xor(ps, m, 32);
            lrow[r] = lrow[r] * a + ps;
            mrow[r] = mnew;
            alpha[r] = a;
            pv0[r] = p0; pv1[r] = p1;
        }
        #pragma unroll
        for (int dt = 0; dt < 8; dt++)
            #pragma unroll
            for (int r = 0; r < 8; r++) Oacc[dt][r] *= alpha[r];

        // ---- repack P (C layout) -> A fragment via per-wave LDS ----
        unsigned short* P = Pt + wv * 16 * PAD32;
        #pragma unroll
        for (int r = 0; r < 8; r++) {
            int row = r + hf * 8;
            P[row * PAD32 + l16]      = f32_to_bf16(pv0[r]);
            P[row * PAD32 + 16 + l16] = f32_to_bf16(pv1[r]);
        }
        Frag16 pf;
        {
            const unsigned short* prow = P + l16 * PAD32;
            #pragma unroll
            for (int v = 0; v < 8; v++) {
                int kpos = hf * 16 + v * 2;
                pf.w[v] = *(const unsigned int*)(prow + kpos);
            }
        }
        // ---- O += P V : 8 d-tiles of 16; V tile is [d][key] so B-frag is contiguous ----
        #pragma unroll
        for (int dt = 0; dt < 8; dt++) {
            Frag16 vf;
            const uint4* base =
                (const uint4*)(&Vt[cur][(dt * 16 + l16) * PAD32 + hf * 16]);
            vf.q[0] = base[0];
            vf.q[1] = base[1];
            Oacc[dt] = __builtin_amdgcn_wmma_f32_16x16x32_bf16(
                false, pf.v, false, vf.v, (short)0, Oacc[dt], false, false);
        }
        // no trailing barrier: next iteration's wait+barrier covers the WAR hazard
    }

    // ---- normalize and store (B,S,HQ*HD) bf16 ----
    #pragma unroll
    for (int dt = 0; dt < 8; dt++) {
        #pragma unroll
        for (int r = 0; r < 8; r++) {
            int s = q0 + r + hf * 8;
            float o = Oacc[dt][r] / lrow[r];
            Oh[((size_t)b * S_ + s) * (HQ_ * HD_) + h * HD_ + dt * 16 + l16] =
                f32_to_bf16(o);
        }
    }
}

// ---------------- host-side orchestration ----------------
extern "C" void kernel_launch(void* const* d_in, const int* in_sizes, int n_in,
                              void* d_out, int out_size, void* d_ws, size_t ws_size,
                              hipStream_t stream) {
    const float* x    = (const float*)d_in[0];
    const float* cosT = (const float*)d_in[1];
    const float* sinT = (const float*)d_in[2];
    const float* Wq   = (const float*)d_in[3];
    const float* bq   = (const float*)d_in[4];
    const float* Wk   = (const float*)d_in[5];
    const float* bk   = (const float*)d_in[6];
    const float* Wv   = (const float*)d_in[7];
    const float* bv   = (const float*)d_in[8];
    const float* Wo   = (const float*)d_in[9];
    // d_in[10..13]: cache_k, cache_v, mask, start_pos — causal/start_pos=0 handled inline

    const size_t M    = (size_t)B_ * S_;             // 4096
    const size_t nQkv = M * (size_t)HQ_ * HD_;       // 16,777,216
    const size_t nKkv = M * (size_t)HKV_ * HD_;      //  4,194,304

    // workspace carve-up (all bf16 as ushort): total ~128 MB
    unsigned short* Qb  = (unsigned short*)d_ws;     // Q proj bf16 [B*S, HQ*HD]   (32 MB)
    unsigned short* Kb  = Qb + nQkv;                 // K proj bf16 [B*S, HKV*HD]  ( 8 MB)
    unsigned short* Vb  = Kb + nKkv;                 // V proj bf16                ( 8 MB)
    unsigned short* Qh  = Vb + nKkv;                 // Q rope (B,HQ,S,HD)         (32 MB)
    unsigned short* Kh  = Qh + nQkv;                 // K rope (B,HKV,S,HD)        ( 8 MB)
    unsigned short* VhT = Kh + nKkv;                 // V transposed (B,HKV,HD,S)  ( 8 MB)
    unsigned short* Oh  = VhT + nKkv;                // attn out (B,S,HQ*HD)       (32 MB)

    dim3 blk(256);

    // 1. QKV projections: stage-convert f32 A/B in-kernel, emit bf16
    gemm_nt_k<float, unsigned short><<<dim3(4096 / 256, 4096 / 128), blk, 0, stream>>>(
        x, Wq, bq, Qb, 4096, 4096, 4096);
    gemm_nt_k<float, unsigned short><<<dim3(1024 / 256, 4096 / 128), blk, 0, stream>>>(
        x, Wk, bk, Kb, 4096, 1024, 4096);
    gemm_nt_k<float, unsigned short><<<dim3(1024 / 256, 4096 / 128), blk, 0, stream>>>(
        x, Wv, bv, Vb, 4096, 1024, 4096);

    // 2. RoPE Q,K + V transpose (independent)
    rope_scatter_k<<<B_ * S_, blk, 0, stream>>>(Qb, Kb, cosT, sinT, Qh, Kh);
    transpose_v_k<<<dim3(S_ / 32, HD_ / 32, B_ * HKV_), blk, 0, stream>>>(Vb, VhT);

    // 3. causal flash attention (async double-buffered K/V staging)
    flash_attn_k<<<dim3(S_ / 128, B_ * HQ_), blk, 0, stream>>>(Qh, Kh, VhT, Oh);

    // 4. output projection: out = attn * Wo^T (bf16 A, f32 weights, f32 out)
    gemm_nt_k<unsigned short, float><<<dim3(4096 / 256, 4096 / 128), blk, 0, stream>>>(
        Oh, Wo, nullptr, (float*)d_out, 4096, 4096, 4096);
}
